// BFLinear_75960791597605
// MI455X (gfx1250) — matrix operand order repaired and checked by reading, not compiled, not run on hardware
//
#include <hip/hip_runtime.h>

typedef __attribute__((ext_vector_type(16))) _Float16 v16h;
typedef __attribute__((ext_vector_type(8)))  float    v8f;

// ---------------------------------------------------------------------------
// Block floating-point quantize (groups of 32 along last dim share exponent,
// 8-bit mantissa). One wave32 == one group. Output exactly representable in
// f16 (q in [-128,127] times power-of-two scale), so downstream f16 WMMA is
// bit-exact w.r.t. the reference's quantized matmul.
// ---------------------------------------------------------------------------
__global__ __launch_bounds__(256) void bfq_f32_to_f16_75960791597605(
    const float* __restrict__ src, _Float16* __restrict__ dst, int n)
{
    int i = blockIdx.x * 256 + threadIdx.x;
    if (i >= n) return;                       // sizes are multiples of 256
    float v = src[i];
    float a = fabsf(v);
    #pragma unroll
    for (int off = 1; off < 32; off <<= 1)    // group max (wave32 == group)
        a = fmaxf(a, __shfl_xor(a, off, 32));
    float q = 0.0f;
    if (a > 0.0f) {
        int e = ilogbf(a);                    // floor(log2(max|x|))
        float s = ldexpf(1.0f, e - 7);        // 2^(e - (BIT-1))
        q = fminf(fmaxf(rintf(v / s), -128.0f), 127.0f) * s;
    }
    dst[i] = (_Float16)q;
}

// ---------------------------------------------------------------------------
// GEMM: out = bfq( xq @ wq^T + 2*b ), f16 inputs, f32 accumulate via
// v_wmma_f32_16x16x32_f16. 256 threads = 8 waves; 128x128 macro tile,
// BK=32 (== one WMMA K == one BFP group). Double-buffered LDS fed by
// gfx1250 GLOBAL_LOAD_ASYNC_TO_LDS_B128 (ASYNCcnt-tracked, no VGPR staging).
// Wave grid 2(M) x 4(N); each wave: 64M x 32N = 4x2 WMMA tiles.
// ---------------------------------------------------------------------------
#define BM 128
#define BN 128
#define BK 32
#define LDT 40   // LDS row stride in halfs (padded: 80B, 16B-aligned)

__global__ __launch_bounds__(256) void bfq_gemm_75960791597605(
    const _Float16* __restrict__ xq,   // (M,K) row-major
    const _Float16* __restrict__ wq,   // (N,K) row-major
    const float* __restrict__ bias,    // (N)
    float* __restrict__ out,           // (M,N)
    int M, int N, int K)
{
    __shared__ _Float16 As[2][BM * LDT];
    __shared__ _Float16 Bs[2][BN * LDT];

    const int t  = threadIdx.x;
    const int m0 = blockIdx.y * BM;
    const int n0 = blockIdx.x * BN;

    // --- async global->LDS tile copy: 4 threads per row, one b128 (8 halfs)
    //     per instruction; each thread covers rows r and r+64 of both tiles.
    const int lrow = t >> 2;            // 0..63
    const int lseg = t & 3;             // 0..3
    const int sOff = lrow * LDT + lseg * 8;

    // per-lane global byte offsets (GVS mode: saddr u64 + vgpr i32 offset)
    const unsigned aOff = (unsigned)(((unsigned)(m0 + lrow) * (unsigned)K
                                      + (unsigned)(lseg * 8)) * 2u);
    const unsigned bOff = (unsigned)(((unsigned)(n0 + lrow) * (unsigned)K
                                      + (unsigned)(lseg * 8)) * 2u);
    const unsigned rowskipB = (unsigned)(64u * (unsigned)K * 2u);

    // wave-relative LDS byte addresses (generic-ptr low 32 bits == LDS offset)
    const unsigned ldsA = (unsigned)(size_t)&As[0][sOff];
    const unsigned ldsB = (unsigned)(size_t)&Bs[0][sOff];
    const unsigned bufSkip  = (unsigned)(BM * LDT * 2);   // As[0] -> As[1]
    const unsigned halfSkip = (unsigned)(64 * LDT * 2);   // row r -> r+64

    auto async_tile = [&](int kt, int buf) {
        const unsigned ga = aOff + (unsigned)(kt * BK * 2);
        const unsigned gb = bOff + (unsigned)(kt * BK * 2);
        const unsigned la = ldsA + (unsigned)buf * bufSkip;
        const unsigned lb = ldsB + (unsigned)buf * bufSkip;
        asm volatile("global_load_async_to_lds_b128 %0, %1, %2"
                     :: "v"(la), "v"(ga), "s"(xq) : "memory");
        asm volatile("global_load_async_to_lds_b128 %0, %1, %2"
                     :: "v"(la + halfSkip), "v"(ga + rowskipB), "s"(xq) : "memory");
        asm volatile("global_load_async_to_lds_b128 %0, %1, %2"
                     :: "v"(lb), "v"(gb), "s"(wq) : "memory");
        asm volatile("global_load_async_to_lds_b128 %0, %1, %2"
                     :: "v"(lb + halfSkip), "v"(gb + rowskipB), "s"(wq) : "memory");
    };

    // --- wave / lane decomposition
    const int wave = t >> 5;
    const int lane = t & 31;
    const int wm = wave >> 2;           // 0..1 : M position (64 rows)
    const int wn = wave & 3;            // 0..3 : N position (32 cols)
    const int lh = lane & 15;
    const int hi = lane >> 4;

    // A frag (16x32 f16): lanes 0-15 row=lh get K {0..7,16..23};
    //                     lanes 16-31 get K {8..15,24..31}  -> kb = hi*8
    const int aBase = (wm * 64 + lh) * LDT + hi * 8;
    // B frag (32x16 f16): lanes 0-15 col=lh get K {0..7,16..23}; 16-31 get K {8..15,24..31}
    const int bBase = (wn * 32 + lh) * LDT + hi * 8;

    v8f acc[4][2] = {};

    const int NK = K / BK;
    async_tile(0, 0);
    asm volatile("s_wait_asynccnt 0x0" ::: "memory");
    __syncthreads();

    for (int kt = 0; kt < NK; ++kt) {
        const int cur = kt & 1;
        // Prefetch next tile straight into the other LDS buffer. Safe: every
        // wave finished READING buf 1-cur at the barrier that ended iter kt-1.
        if (kt + 1 < NK) async_tile(kt + 1, 1 - cur);

        union { v16h v; uint4 q[2]; } af[4], bf[2];
        #pragma unroll
        for (int i = 0; i < 4; ++i) {
            const _Float16* p = &As[cur][aBase + i * 16 * LDT];
            af[i].q[0] = *(const uint4*)p;         // K 0..7  (+kb)
            af[i].q[1] = *(const uint4*)(p + 16);  // K 16..23 (+kb)
        }
        #pragma unroll
        for (int j = 0; j < 2; ++j) {
            const _Float16* p = &Bs[cur][bBase + j * 16 * LDT];
            bf[j].q[0] = *(const uint4*)p;         // K kb..kb+7
            bf[j].q[1] = *(const uint4*)(p + 16);  // K 16+kb..16+kb+7
        }
        #pragma unroll
        for (int i = 0; i < 4; ++i)
            #pragma unroll
            for (int j = 0; j < 2; ++j)
                acc[i][j] = __builtin_amdgcn_wmma_f32_16x16x32_f16(
                    false, af[i].v, false, bf[j].v,
                    (short)0, acc[i][j], false, false);

        if (kt + 1 < NK) {
            asm volatile("s_wait_asynccnt 0x0" ::: "memory");
            __syncthreads();
        }
    }

    // --- epilogue: +2*b, then BF-quantize per 32-col group.
    // A wave's 32 N-columns (aligned to 32) form exactly one quant group.
    // C layout: VGPR r -> row (r + 8*hi), lane lh -> col.
    const int nG = n0 + wn * 32 + lh;
    const float b0 = 2.0f * bias[nG];
    const float b1 = 2.0f * bias[nG + 16];

    #pragma unroll
    for (int i = 0; i < 4; ++i) {
        const int mBase = m0 + wm * 64 + i * 16 + hi * 8;
        #pragma unroll
        for (int r = 0; r < 8; ++r) {
            float v0 = acc[i][0][r] + b0;
            float v1 = acc[i][1][r] + b1;
            float am = fmaxf(fabsf(v0), fabsf(v1));
            #pragma unroll
            for (int off = 1; off < 16; off <<= 1)   // reduce across 16 cols
                am = fmaxf(am, __shfl_xor(am, off, 16));
            float q0 = 0.0f, q1 = 0.0f;
            if (am > 0.0f) {
                int e = ilogbf(am);
                float s = ldexpf(1.0f, e - 7);
                float inv = 1.0f / s;                // power of two: exact
                q0 = fminf(fmaxf(rintf(v0 * inv), -128.0f), 127.0f) * s;
                q1 = fminf(fmaxf(rintf(v1 * inv), -128.0f), 127.0f) * s;
            }
            const size_t o = (size_t)(mBase + r) * N + nG;
            out[o]      = q0;
            out[o + 16] = q1;
        }
    }
}

// ---------------------------------------------------------------------------
extern "C" void kernel_launch(void* const* d_in, const int* in_sizes, int n_in,
                              void* d_out, int out_size, void* d_ws, size_t ws_size,
                              hipStream_t stream) {
    const float* x = (const float*)d_in[0];   // (M, K) f32
    const float* W = (const float*)d_in[1];   // (N, K) f32
    const float* b = (const float*)d_in[2];   // (N)    f32
    float* out = (float*)d_out;

    const int nx = in_sizes[0];               // M*K
    const int nw = in_sizes[1];               // N*K
    const int N  = in_sizes[2];               // 4096
    const int K  = nw / N;                    // 4096
    const int M  = nx / K;                    // 8192

    _Float16* xq = (_Float16*)d_ws;           // M*K f16  (64 MB)
    _Float16* wq = xq + (size_t)nx;           // N*K f16  (32 MB)

    bfq_f32_to_f16_75960791597605<<<(nx + 255) / 256, 256, 0, stream>>>(x, xq, nx);
    bfq_f32_to_f16_75960791597605<<<(nw + 255) / 256, 256, 0, stream>>>(W, wq, nw);

    dim3 grid(N / BN, M / BM);                // (32, 64)
    bfq_gemm_75960791597605<<<grid, 256, 0, stream>>>(xq, wq, b, out, M, N, K);
}
